// BiLSTMCRF_90305982365833
// MI455X (gfx1250) — matrix operand order repaired and checked by reading, not compiled, not run on hardware
//
#include <hip/hip_runtime.h>
#include <hip/hip_bf16.h>

// BiLSTM-CRF forward loss for MI455X (gfx1250), wave32 + v_wmma_f32_16x16x32_f16.
// All GEMMs (char/word LSTM projections+recurrence, tag head) run on the WMMA path.
// Main GEMM: 16Mx64N tile per wave -> 4 independent WMMA accumulator chains per
// K-step (hides XDL latency), A fragment loaded once per K-step (4x reuse).

typedef _Float16 v8h  __attribute__((ext_vector_type(8)));
typedef _Float16 v16h __attribute__((ext_vector_type(16)));
typedef float    v8f  __attribute__((ext_vector_type(8)));

#define B_    32
#define S_    256
#define CW_   12
#define HC_   128
#define HID_  512
#define TAGS_ 20
#define START_ 18
#define STOP_  19

union V16u { v16h v; v8h h[2]; };

// ---------------------------------------------------------------------------
// Operand layouts as loaded here (wave32; contiguous K in A and B, not the ISA table order):
//  A (16x32 f16): lane = row(M)&15; halves 0..15 = K kb..kb+15,
//    kb = 16*(lane>=16)  -> two 16B loads.
//  B (32x16 f16, from row-major W[n][k]): lane = col(N)&15; 16 contiguous K
//    halves starting at 16*(lane>=16)     -> one 32B load.
//  C/D (16x16 f32): lane holds col = lane&15; vgpr v holds row = v+8*(lane>=16).
// ---------------------------------------------------------------------------

// Wide GEMM: C[M,N](f32,ldc) = A[M,K](f16,lda) * W[N,K](f16,ldw)^T
// Requires M%16==0, N%64==0, K%32==0. One wave owns a 16x64 tile.
__global__ void gemm_wmma_n64_kernel(const _Float16* __restrict__ A, long lda,
                                     const _Float16* __restrict__ W, long ldw,
                                     float* __restrict__ C, long ldc,
                                     int M, int N, int K) {
  int wavesPerBlock = blockDim.x >> 5;
  int wave = blockIdx.x * wavesPerBlock + ((int)threadIdx.x >> 5);
  int lane = (int)threadIdx.x & 31;
  int tilesN = N >> 6;
  int totalTiles = (M >> 4) * tilesN;
  if (wave >= totalTiles) return;          // wave-uniform: EXEC stays all-1s
  int tm = wave / tilesN;
  int tn = wave - tm * tilesN;
  int half = lane >> 4;                    // 0 or 1
  int r = lane & 15;

  const _Float16* arow  = A + (long)(tm * 16 + r) * lda + half * 16;
  const _Float16* wrow0 = W + (long)(tn * 64 + r) * ldw + half * 16;
  const long wstep = 16 * ldw;

  v8f acc0 = {0.f,0.f,0.f,0.f,0.f,0.f,0.f,0.f};
  v8f acc1 = acc0, acc2 = acc0, acc3 = acc0;

  for (int k0 = 0; k0 < K; k0 += 32) {
    V16u a;
    a.h[0] = *(const v8h*)(arow + k0);
    a.h[1] = *(const v8h*)(arow + k0 + 8);
    v16h b0 = *(const v16h*)(wrow0 + k0);
    v16h b1 = *(const v16h*)(wrow0 + wstep + k0);
    v16h b2 = *(const v16h*)(wrow0 + 2 * wstep + k0);
    v16h b3 = *(const v16h*)(wrow0 + 3 * wstep + k0);
    acc0 = __builtin_amdgcn_wmma_f32_16x16x32_f16(false, a.v, false, b0,
                                                  (short)0, acc0, false, false);
    acc1 = __builtin_amdgcn_wmma_f32_16x16x32_f16(false, a.v, false, b1,
                                                  (short)0, acc1, false, false);
    acc2 = __builtin_amdgcn_wmma_f32_16x16x32_f16(false, a.v, false, b2,
                                                  (short)0, acc2, false, false);
    acc3 = __builtin_amdgcn_wmma_f32_16x16x32_f16(false, a.v, false, b3,
                                                  (short)0, acc3, false, false);
  }

  int rbase = tm * 16 + half * 8;
  int cbase = tn * 64 + r;
  float* crow = C + (long)rbase * ldc + cbase;
#pragma unroll
  for (int v = 0; v < 8; ++v) {
    crow[(long)v * ldc]      = acc0[v];
    crow[(long)v * ldc + 16] = acc1[v];
    crow[(long)v * ldc + 32] = acc2[v];
    crow[(long)v * ldc + 48] = acc3[v];
  }
}

// Generic 16x16-tile GEMM with masked N store (used for the 20-tag head).
__global__ void gemm_wmma_kernel(const _Float16* __restrict__ A, long lda,
                                 const _Float16* __restrict__ W, long ldw,
                                 float* __restrict__ C, long ldc,
                                 int M, int N, int K, int nMax) {
  int wavesPerBlock = blockDim.x >> 5;
  int wave = blockIdx.x * wavesPerBlock + ((int)threadIdx.x >> 5);
  int lane = (int)threadIdx.x & 31;
  int tilesN = N >> 4;
  int totalTiles = (M >> 4) * tilesN;
  if (wave >= totalTiles) return;
  int tm = wave / tilesN;
  int tn = wave - tm * tilesN;
  int half = lane >> 4;
  int r = lane & 15;

  const _Float16* arow = A + (long)(tm * 16 + r) * lda + half * 16;
  const _Float16* wrow = W + (long)(tn * 16 + r) * ldw + half * 16;

  v8f acc = {0.f,0.f,0.f,0.f,0.f,0.f,0.f,0.f};
  for (int k0 = 0; k0 < K; k0 += 32) {
    V16u a, b;
    a.h[0] = *(const v8h*)(arow + k0);
    a.h[1] = *(const v8h*)(arow + k0 + 8);
    b.v    = *(const v16h*)(wrow + k0);
    acc = __builtin_amdgcn_wmma_f32_16x16x32_f16(false, a.v, false, b.v,
                                                 (short)0, acc, false, false);
  }

  int cn = tn * 16 + r;
  int rbase = tm * 16 + half * 8;
  if (cn < nMax) {
#pragma unroll
    for (int v = 0; v < 8; ++v)
      C[(long)(rbase + v) * ldc + cn] = acc[v];
  }
}

// ---------------------------------------------------------------------------
// Elementwise / conversion kernels
// ---------------------------------------------------------------------------
__global__ void f32_to_f16_kernel(const float* __restrict__ src,
                                  _Float16* __restrict__ dst, long n) {
  long i = (long)blockIdx.x * blockDim.x + threadIdx.x;
  if (i < n) dst[i] = (_Float16)src[i];
}

// h2t_W [20,1024] f32 -> padded [32,1024] f16 (rows 20..31 zero)
__global__ void h2t_pad_kernel(const float* __restrict__ src,
                               _Float16* __restrict__ dst) {
  int i = blockIdx.x * blockDim.x + threadIdx.x;
  if (i >= 32 * 1024) return;
  int r = i >> 10, c = i & 1023;
  dst[i] = (r < TAGS_) ? (_Float16)src[r * 1024 + c] : (_Float16)0.f;
}

// chars [S*B*C] -> ce f16 [S*B*C, 128]
__global__ void gather_char_kernel(const int* __restrict__ chars,
                                   const float* __restrict__ emb,
                                   _Float16* __restrict__ dst, long n) {
  long i = (long)blockIdx.x * blockDim.x + threadIdx.x;
  if (i >= n) return;
  long tok = i >> 7;
  int h = (int)(i & 127);
  dst[i] = (_Float16)emb[(long)chars[tok] * HC_ + h];
}

// x[b,s,0:768] = [hT_fwd(128) | hT_bwd(128) | word_emb(512)]  (f16)
__global__ void build_x_kernel(const _Float16* __restrict__ hcf,
                               const _Float16* __restrict__ hcb,
                               const int* __restrict__ words,
                               const float* __restrict__ wemb,
                               _Float16* __restrict__ x) {
  long i = (long)blockIdx.x * blockDim.x + threadIdx.x;
  if (i >= (long)B_ * S_ * 768) return;
  long rrow = i / 768;
  int col = (int)(i % 768);
  int b = (int)(rrow >> 8), s = (int)(rrow & 255);
  long sb = (long)s * B_ + b;            // char rows are (s*B+b)
  _Float16 v;
  if (col < 128)      v = hcf[sb * 128 + col];
  else if (col < 256) v = hcb[sb * 128 + (col - 128)];
  else                v = (_Float16)wemb[(long)words[rrow] * HID_ + (col - 256)];
  x[i] = v;
}

__device__ __forceinline__ float sigf(float x) { return 1.f / (1.f + expf(-x)); }

// LSTM cell (PyTorch gate order i,f,g,o). pre = gIn + bias (+ gRec if !first).
// Writes c (f32 state) and h (f16) at hOut[n*hStride + j].
__global__ void lstm_cell_kernel(const float* __restrict__ gIn, long gInStride,
                                 const float* __restrict__ gRec,
                                 const float* __restrict__ bias,
                                 float* __restrict__ c,
                                 _Float16* __restrict__ hOut, long hStride,
                                 int N, int H, int first) {
  long i = (long)blockIdx.x * blockDim.x + threadIdx.x;
  if (i >= (long)N * H) return;
  int n = (int)(i / H), j = (int)(i % H);
  const float* gi = gIn + (long)n * gInStride;
  float pi = gi[j]         + bias[j];
  float pf = gi[H + j]     + bias[H + j];
  float pg = gi[2 * H + j] + bias[2 * H + j];
  float po = gi[3 * H + j] + bias[3 * H + j];
  if (!first) {
    const float* gr = gRec + (long)n * 4 * H;
    pi += gr[j]; pf += gr[H + j]; pg += gr[2 * H + j]; po += gr[3 * H + j];
  }
  float cp = first ? 0.f : c[i];
  float cn = sigf(pf) * cp + sigf(pi) * tanhf(pg);
  float hn = sigf(po) * tanhf(cn);
  c[i] = cn;
  hOut[(long)n * hStride + j] = (_Float16)hn;
}

// ---------------------------------------------------------------------------
// CRF: forward partition (log-sum-exp recurrence) + gold path score + loss.
// Single block, 1024 threads: b = tid>>5 (32 batches), t = tid&31 (t<20 live).
// ---------------------------------------------------------------------------
__global__ void crf_kernel(const float* __restrict__ logits,   // [B*S, 20]
                           const float* __restrict__ h2tb,     // [20]
                           const float* __restrict__ trans,    // [20,20]
                           const int* __restrict__ tags,       // [B,S]
                           const unsigned char* __restrict__ mask, // [B,S]
                           float* __restrict__ out) {
  __shared__ float alpha[B_][TAGS_];
  __shared__ float nalpha[B_][TAGS_];
  __shared__ float tr[TAGS_ * TAGS_];
  __shared__ float fsc[B_], gsc[B_];
  int tid = (int)threadIdx.x;
  int b = tid >> 5, t = tid & 31;
  if (tid < TAGS_ * TAGS_) tr[tid] = trans[tid];
  if (t < TAGS_) alpha[b][t] = (t == START_) ? 0.f : -10000.f;
  __syncthreads();

  for (int s = 0; s < S_; ++s) {
    if (t < TAGS_) {
      float cur = logits[((long)b * S_ + s) * TAGS_ + t] + h2tb[t];
      float m = -3.4e38f;
      for (int j = 0; j < TAGS_; ++j)
        m = fmaxf(m, alpha[b][j] + tr[t * TAGS_ + j]);
      float sum = 0.f;
      for (int j = 0; j < TAGS_; ++j)
        sum += expf(alpha[b][j] + tr[t * TAGS_ + j] - m);
      nalpha[b][t] = m + logf(sum) + cur;
    }
    __syncthreads();
    if (t < TAGS_) alpha[b][t] = nalpha[b][t];
    __syncthreads();
  }

  if (t == 0) {
    float m = -3.4e38f;
    for (int j = 0; j < TAGS_; ++j)
      m = fmaxf(m, alpha[b][j] + tr[STOP_ * TAGS_ + j]);
    float sum = 0.f;
    for (int j = 0; j < TAGS_; ++j)
      sum += expf(alpha[b][j] + tr[STOP_ * TAGS_ + j] - m);
    fsc[b] = m + logf(sum);

    float g = 0.f;
    int prev = START_;
    for (int s = 0; s < S_; ++s) {
      int tg = tags[b * S_ + s];
      float sc = tr[tg * TAGS_ + prev] +
                 logits[((long)b * S_ + s) * TAGS_ + tg] + h2tb[tg];
      if (mask[b * S_ + s]) sc = 0.f;
      g += sc;
      prev = tg;
    }
    g += tr[STOP_ * TAGS_ + tags[b * S_ + (S_ - 1)]];
    gsc[b] = g;
  }
  __syncthreads();
  if (tid == 0) {
    float acc = 0.f;
    for (int bb = 0; bb < B_; ++bb) acc += fsc[bb] - gsc[bb];  // mean(f - g)
    out[0] = acc / (float)B_;
  }
}

// ---------------------------------------------------------------------------
// Host orchestration
// ---------------------------------------------------------------------------
static inline void launch_gemm(const _Float16* A, long lda, const _Float16* W,
                               long ldw, float* C, long ldc, int M, int N,
                               int K, int nMax, hipStream_t stream) {
  if ((N & 63) == 0 && nMax == N) {
    int tiles = (M >> 4) * (N >> 6);
    int blocks = (tiles + 7) / 8;  // 8 waves (256 threads) per block
    gemm_wmma_n64_kernel<<<blocks, 256, 0, stream>>>(A, lda, W, ldw, C, ldc, M, N, K);
  } else {
    int tiles = (M >> 4) * (N >> 4);
    int blocks = (tiles + 7) / 8;
    gemm_wmma_kernel<<<blocks, 256, 0, stream>>>(A, lda, W, ldw, C, ldc, M, N, K, nMax);
  }
}

static inline int gsz(long n) { return (int)((n + 255) / 256); }

extern "C" void kernel_launch(void* const* d_in, const int* in_sizes, int n_in,
                              void* d_out, int out_size, void* d_ws, size_t ws_size,
                              hipStream_t stream) {
  (void)in_sizes; (void)n_in; (void)out_size; (void)ws_size;
  // Inputs in setup_inputs() flattening order:
  const float* word_emb = (const float*)d_in[0];   // [50000,512]
  const float* char_emb = (const float*)d_in[1];   // [128,128]
  const float* cfW = (const float*)d_in[2];        // [512,128]
  const float* cfU = (const float*)d_in[3];        // [512,128]
  const float* cfb = (const float*)d_in[4];        // [512]
  const float* cbW = (const float*)d_in[5];
  const float* cbU = (const float*)d_in[6];
  const float* cbb = (const float*)d_in[7];
  const float* wfW = (const float*)d_in[8];        // [2048,768]
  const float* wfU = (const float*)d_in[9];        // [2048,512]
  const float* wfb = (const float*)d_in[10];       // [2048]
  const float* wbW = (const float*)d_in[11];
  const float* wbU = (const float*)d_in[12];
  const float* wbb = (const float*)d_in[13];
  const float* h2tW = (const float*)d_in[14];      // [20,1024]
  const float* h2tb = (const float*)d_in[15];      // [20]
  const float* trans = (const float*)d_in[16];     // [20,20]
  const int* words = (const int*)d_in[17];         // [32,256]
  const int* chars = (const int*)d_in[18];         // [256,32,12]
  const int* tags = (const int*)d_in[19];          // [32,256]
  const unsigned char* mask = (const unsigned char*)d_in[20];  // [32,256] bool

  const long NCH = (long)S_ * B_;       // 8192 char sequences
  const long NTOK = NCH * CW_;          // 98304 char tokens

  // bump allocator over workspace
  char* base = (char*)d_ws;
  size_t off = 0;
  auto alloc = [&](size_t bytes) -> void* {
    off = (off + 255) & ~(size_t)255;
    void* p = (void*)(base + off);
    off += bytes;
    return p;
  };

  _Float16* ce    = (_Float16*)alloc(NTOK * HC_ * 2);            // [98304,128] f16
  _Float16* cfW16 = (_Float16*)alloc(512 * 128 * 2);
  _Float16* cfU16 = (_Float16*)alloc(512 * 128 * 2);
  _Float16* cbW16 = (_Float16*)alloc(512 * 128 * 2);
  _Float16* cbU16 = (_Float16*)alloc(512 * 128 * 2);
  _Float16* wfW16 = (_Float16*)alloc((size_t)2048 * 768 * 2);
  _Float16* wbW16 = (_Float16*)alloc((size_t)2048 * 768 * 2);
  _Float16* wfU16 = (_Float16*)alloc((size_t)2048 * 512 * 2);
  _Float16* wbU16 = (_Float16*)alloc((size_t)2048 * 512 * 2);
  _Float16* h2t16 = (_Float16*)alloc((size_t)32 * 1024 * 2);     // padded 20->32 rows
  _Float16* hcf   = (_Float16*)alloc(NCH * HC_ * 2);             // char hT fwd
  _Float16* hcb   = (_Float16*)alloc(NCH * HC_ * 2);             // char hT bwd
  float* cst_c    = (float*)alloc(NCH * HC_ * 4);                // char c state
  float* gIn_c    = (float*)alloc(NCH * 512 * 4);                // char input gates
  float* gRec_c   = (float*)alloc(NCH * 512 * 4);                // char recur gates
  _Float16* xw    = (_Float16*)alloc((size_t)B_ * S_ * 768 * 2); // word input f16
  float* gIn_w    = (float*)alloc((size_t)B_ * 2048 * 4);
  float* gRec_w   = (float*)alloc((size_t)B_ * 2048 * 4);
  float* cst_w    = (float*)alloc((size_t)B_ * HID_ * 4);
  _Float16* hs    = (_Float16*)alloc((size_t)B_ * S_ * 1024 * 2); // [B,S,2*HID] f16
  float* logits   = (float*)alloc((size_t)B_ * S_ * TAGS_ * 4);

  // --- weight / embedding conversion ---
  f32_to_f16_kernel<<<gsz(512 * 128), 256, 0, stream>>>(cfW, cfW16, 512 * 128);
  f32_to_f16_kernel<<<gsz(512 * 128), 256, 0, stream>>>(cfU, cfU16, 512 * 128);
  f32_to_f16_kernel<<<gsz(512 * 128), 256, 0, stream>>>(cbW, cbW16, 512 * 128);
  f32_to_f16_kernel<<<gsz(512 * 128), 256, 0, stream>>>(cbU, cbU16, 512 * 128);
  f32_to_f16_kernel<<<gsz((long)2048 * 768), 256, 0, stream>>>(wfW, wfW16, (long)2048 * 768);
  f32_to_f16_kernel<<<gsz((long)2048 * 768), 256, 0, stream>>>(wbW, wbW16, (long)2048 * 768);
  f32_to_f16_kernel<<<gsz((long)2048 * 512), 256, 0, stream>>>(wfU, wfU16, (long)2048 * 512);
  f32_to_f16_kernel<<<gsz((long)2048 * 512), 256, 0, stream>>>(wbU, wbU16, (long)2048 * 512);
  h2t_pad_kernel<<<gsz(32 * 1024), 256, 0, stream>>>(h2tW, h2t16);
  gather_char_kernel<<<gsz(NTOK * HC_), 256, 0, stream>>>(chars, char_emb, ce, NTOK * HC_);

  // --- char BiLSTM: N=8192, T=12, H=128; keep only final hidden per direction ---
  {
    const _Float16* Wih[2] = {cfW16, cbW16};
    const _Float16* Whh[2] = {cfU16, cbU16};
    const float* bias[2] = {cfb, cbb};
    _Float16* hst[2] = {hcf, hcb};
    for (int d = 0; d < 2; ++d) {
      for (int t = 0; t < CW_; ++t) {
        int tc = d ? (CW_ - 1 - t) : t;
        // input projection for this timestep: [8192,128] @ [512,128]^T
        launch_gemm(ce + (long)tc * HC_, (long)CW_ * HC_, Wih[d], HC_,
                    gIn_c, 512, (int)NCH, 512, HC_, 512, stream);
        if (t > 0)
          launch_gemm(hst[d], HC_, Whh[d], HC_, gRec_c, 512,
                      (int)NCH, 512, HC_, 512, stream);
        long n = NCH * HC_;
        lstm_cell_kernel<<<gsz(n), 256, 0, stream>>>(
            gIn_c, 512, (t > 0) ? gRec_c : nullptr, bias[d],
            cst_c, hst[d], HC_, (int)NCH, HC_, (t == 0) ? 1 : 0);
      }
    }
  }

  // --- word input: concat(h_char_fwd, h_char_bwd, word_emb) -> f16 [B,S,768] ---
  build_x_kernel<<<gsz((long)B_ * S_ * 768), 256, 0, stream>>>(hcf, hcb, words, word_emb, xw);

  // --- word BiLSTM: N=32, T=256, H=512; h written straight into hs (f16) ---
  {
    const _Float16* Wih[2] = {wfW16, wbW16};
    const _Float16* Whh[2] = {wfU16, wbU16};
    const float* bias[2] = {wfb, wbb};
    const long ldx = (long)S_ * 768;     // batch stride in x
    const long ldh = (long)S_ * 1024;    // batch stride in hs
    for (int d = 0; d < 2; ++d) {
      for (int t = 0; t < S_; ++t) {
        int tc = d ? (S_ - 1 - t) : t;
        launch_gemm(xw + (long)tc * 768, ldx, Wih[d], 768,
                    gIn_w, 2048, B_, 2048, 768, 2048, stream);
        if (t > 0) {
          int tp = d ? (tc + 1) : (tc - 1);  // previously-computed step
          launch_gemm(hs + (long)tp * 1024 + d * HID_, ldh, Whh[d], HID_,
                      gRec_w, 2048, B_, 2048, HID_, 2048, stream);
        }
        long n = (long)B_ * HID_;
        lstm_cell_kernel<<<gsz(n), 256, 0, stream>>>(
            gIn_w, 2048, (t > 0) ? gRec_w : nullptr, bias[d],
            cst_w, hs + (long)tc * 1024 + d * HID_, ldh, B_, HID_, (t == 0) ? 1 : 0);
      }
    }
  }

  // --- tag logits: [8192,1024] @ [20(pad 32),1024]^T, masked store to [.,20] ---
  launch_gemm(hs, 1024, h2t16, 1024, logits, TAGS_, (int)((long)B_ * S_), 32, 1024,
              TAGS_, stream);

  // --- CRF partition + gold score + mean loss ---
  crf_kernel<<<1, 1024, 0, stream>>>(logits, h2tb, trans, tags, mask, (float*)d_out);
}
